// self_attn_79233556676625
// MI455X (gfx1250) — matrix operand order repaired and checked by reading, not compiled, not run on hardware
//
#include <hip/hip_runtime.h>
#include <hip/hip_bf16.h>
#include <stdint.h>

// Self-attention forward on gfx1250: cooperative 128x128 block tiles,
// double-buffered LDS staging via global_load_async_to_lds_b128 (ASYNCcnt),
// compute on v_wmma_f32_16x16x32_bf16.
// ws layout (bf16 elems): Wb[3*512*512] | Qb[B*S*D] | Kb[B*S*D] | Vt[B*D*S] (~25.7MB)

typedef __bf16 bf16;
typedef __attribute__((ext_vector_type(16))) __bf16 v16bf;
typedef __attribute__((ext_vector_type(8)))  __bf16 v8bf;
typedef __attribute__((ext_vector_type(8)))  float  v8f;

#define B_  4
#define S_  2048
#define DI_ 512
#define DQ_ 512

#define LDS_STRIDE 40   // 32 bf16 data + 8 pad: rows stay 16B aligned, banks spread

// ---------------- WMMA + layout helpers (CDNA5 ISA 7.12.2) ----------------

__device__ __forceinline__ v8f wmma_bf16(v16bf a, v16bf b, v8f c) {
  return __builtin_amdgcn_wmma_f32_16x16x32_bf16(false, a, false, b,
                                                 (short)0, c, false, false);
}

// A 16x32 bf16 tile from LDS: lane half=lane>>4, m=lane&15;
// chunks at K = 8*half and K = 16 + 8*half.
__device__ __forceinline__ v16bf lds_a(const bf16* base, int m0, int lane) {
  int hf = lane >> 4, m = lane & 15;
  const bf16* r = base + (m0 + m) * LDS_STRIDE + 8 * hf;
  v8bf lo = *(const v8bf*)(r);
  v8bf hi = *(const v8bf*)(r + 16);
  return __builtin_shufflevector(lo, hi, 0,1,2,3,4,5,6,7,8,9,10,11,12,13,14,15);
}

// B 32x16 bf16 tile from LDS [n][k]: lane n=lane&15; chunks at K = 8*half and K = 16 + 8*half.
__device__ __forceinline__ v16bf lds_b(const bf16* base, int n0, int lane) {
  int hf = lane >> 4, n = lane & 15;
  const bf16* r = base + (n0 + n) * LDS_STRIDE + 8 * hf;
  v8bf lo = *(const v8bf*)(r);
  v8bf hi = *(const v8bf*)(r + 16);
  return __builtin_shufflevector(lo, hi, 0,1,2,3,4,5,6,7,8,9,10,11,12,13,14,15);
}

// ---------------- staging: global -> LDS ----------------

// 128x32 bf16 tile (row-major src, ld elems) via async-to-LDS. 512 x 16B chunks,
// 2 per thread. Tracked by ASYNCcnt.
__device__ __forceinline__ void stage_async_bf16(bf16* lds, const bf16* src,
                                                 int ld, int row0, int k0, int tid) {
#pragma unroll
  for (int i = 0; i < 2; ++i) {
    int idx = tid + i * 256;
    int r = idx >> 2, c = idx & 3;
    const bf16* g = src + (size_t)(row0 + r) * ld + (k0 + c * 8);
    uint32_t l = (uint32_t)(uintptr_t)(lds + r * LDS_STRIDE + c * 8);
    asm volatile("global_load_async_to_lds_b128 %0, %1, off"
                 :: "v"(l), "v"(g) : "memory");
  }
}

// 128x32 tile from f32 source: load, cvt to bf16, ds_store (DScnt path).
__device__ __forceinline__ void stage_cvt_f32(bf16* lds, const float* src,
                                              int ld, int row0, int k0, int tid) {
#pragma unroll
  for (int i = 0; i < 2; ++i) {
    int idx = tid + i * 256;
    int r = idx >> 2, c = idx & 3;
    v8f f = *(const v8f*)(src + (size_t)(row0 + r) * ld + (k0 + c * 8));
    v8bf h;
#pragma unroll
    for (int j = 0; j < 8; ++j) h[j] = (bf16)f[j];
    *(v8bf*)(lds + r * LDS_STRIDE + c * 8) = h;
  }
}

__device__ __forceinline__ void wait_async() {
  asm volatile("s_wait_asynccnt 0x0" ::: "memory");
}

// ---------------- Kernel 0: convert weights to bf16 ----------------
__global__ __launch_bounds__(256) void convert_w_kernel(
    const float* __restrict__ Wq, const float* __restrict__ Wk,
    const float* __restrict__ Wv, bf16* __restrict__ Wb) {
  const int N = DQ_ * DI_;
  int i = blockIdx.x * 256 + threadIdx.x;
  if (i < N) {
    Wb[i]         = (bf16)Wq[i];
    Wb[N + i]     = (bf16)Wk[i];
    Wb[2 * N + i] = (bf16)Wv[i];
  }
}

// ---------------- Kernel 1: Q/K/V projections (x f32, W bf16) ----------------
// grid: mat(3) x Mblk(8192/128=64) x Nblk(512/128=4) = 768 blocks
__global__ __launch_bounds__(256) void qkv_kernel(
    const float* __restrict__ x, const bf16* __restrict__ Wb,
    const float* __restrict__ bq, const float* __restrict__ bk,
    const float* __restrict__ bv,
    bf16* __restrict__ Qb, bf16* __restrict__ Kb, bf16* __restrict__ Vt) {
  __shared__ bf16 As[2][128 * LDS_STRIDE];
  __shared__ bf16 Bs[2][128 * LDS_STRIDE];

  int tid = threadIdx.x, lane = tid & 31, wv = tid >> 5;
  int wr = wv >> 1, wc = wv & 1;

  int bid  = blockIdx.x;
  int mat  = bid >> 8;            // 0=Q 1=K 2=V
  int rem  = bid & 255;
  int row0 = (rem >> 2) * 128;    // over B*S
  int n0   = (rem & 3) * 128;

  const bf16* W = Wb + (size_t)mat * DQ_ * DI_;

  v8f acc[2][4];
#pragma unroll
  for (int mi = 0; mi < 2; ++mi)
#pragma unroll
    for (int t = 0; t < 4; ++t) acc[mi][t] = v8f{};

  stage_cvt_f32(As[0], x, DI_, row0, 0, tid);
  stage_async_bf16(Bs[0], W, DI_, n0, 0, tid);

  int buf = 0;
  for (int k0 = 0; k0 < DI_; k0 += 32) {
    wait_async();
    __syncthreads();
    int nxt = buf ^ 1;
    if (k0 + 32 < DI_) {
      stage_cvt_f32(As[nxt], x, DI_, row0, k0 + 32, tid);
      stage_async_bf16(Bs[nxt], W, DI_, n0, k0 + 32, tid);
    }
    v16bf a0 = lds_a(As[buf], wr * 32, lane);
    v16bf a1 = lds_a(As[buf], wr * 32 + 16, lane);
#pragma unroll
    for (int t = 0; t < 4; ++t) {
      v16bf b = lds_b(Bs[buf], wc * 64 + t * 16, lane);
      acc[0][t] = wmma_bf16(a0, b, acc[0][t]);
      acc[1][t] = wmma_bf16(a1, b, acc[1][t]);
    }
    __syncthreads();
    buf = nxt;
  }

  int hf = lane >> 4, n = lane & 15;
  int mbase = row0 + wr * 32, nbase = n0 + wc * 64;
  if (mat < 2) {
    bf16* O = (mat == 0) ? Qb : Kb;
    const float* bias = (mat == 0) ? bq : bk;
#pragma unroll
    for (int mi = 0; mi < 2; ++mi)
#pragma unroll
      for (int t = 0; t < 4; ++t) {
        int col = nbase + t * 16 + n; float bb = bias[col];
#pragma unroll
        for (int r = 0; r < 8; ++r)
          O[(size_t)(mbase + mi * 16 + hf * 8 + r) * DQ_ + col] =
              (bf16)(acc[mi][t][r] + bb);
      }
  } else {  // V stored transposed: Vt[b][d][t]
    int bidx = row0 / S_, sbase = (row0 % S_) + wr * 32;
    bf16* Vb = Vt + (size_t)bidx * DQ_ * S_;
#pragma unroll
    for (int mi = 0; mi < 2; ++mi)
#pragma unroll
      for (int t = 0; t < 4; ++t) {
        int col = nbase + t * 16 + n; float bb = bv[col];
#pragma unroll
        for (int r = 0; r < 8; ++r)
          Vb[(size_t)col * S_ + sbase + mi * 16 + hf * 8 + r] =
              (bf16)(acc[mi][t][r] + bb);
      }
  }
}

// ---------------- Kernel 2: scores = Q @ K^T * scale ----------------
// grid: b(4) x Sblk(16) x Tblk(16) = 1024 blocks
__global__ __launch_bounds__(256) void scores_kernel(
    const bf16* __restrict__ Qb, const bf16* __restrict__ Kb,
    float* __restrict__ attn) {
  __shared__ bf16 As[2][128 * LDS_STRIDE];
  __shared__ bf16 Bs[2][128 * LDS_STRIDE];

  int tid = threadIdx.x, lane = tid & 31, wv = tid >> 5;
  int wr = wv >> 1, wc = wv & 1;

  int bid = blockIdx.x;
  int b   = bid >> 8;
  int rem = bid & 255;
  int s0  = (rem >> 4) * 128;
  int t0  = (rem & 15) * 128;

  const bf16* Q = Qb + (size_t)b * S_ * DQ_;
  const bf16* K = Kb + (size_t)b * S_ * DQ_;

  v8f acc[2][4];
#pragma unroll
  for (int mi = 0; mi < 2; ++mi)
#pragma unroll
    for (int t = 0; t < 4; ++t) acc[mi][t] = v8f{};

  stage_async_bf16(As[0], Q, DQ_, s0, 0, tid);
  stage_async_bf16(Bs[0], K, DQ_, t0, 0, tid);

  int buf = 0;
  for (int k0 = 0; k0 < DQ_; k0 += 32) {
    wait_async();
    __syncthreads();
    int nxt = buf ^ 1;
    if (k0 + 32 < DQ_) {
      stage_async_bf16(As[nxt], Q, DQ_, s0, k0 + 32, tid);
      stage_async_bf16(Bs[nxt], K, DQ_, t0, k0 + 32, tid);
    }
    v16bf a0 = lds_a(As[buf], wr * 32, lane);
    v16bf a1 = lds_a(As[buf], wr * 32 + 16, lane);
#pragma unroll
    for (int t = 0; t < 4; ++t) {
      v16bf bm = lds_b(Bs[buf], wc * 64 + t * 16, lane);
      acc[0][t] = wmma_bf16(a0, bm, acc[0][t]);
      acc[1][t] = wmma_bf16(a1, bm, acc[1][t]);
    }
    __syncthreads();
    buf = nxt;
  }

  const float scale = 0.04419417382415922f;  // 1/sqrt(512)
  int hf = lane >> 4, n = lane & 15;
  float* out = attn + (size_t)b * S_ * S_;
#pragma unroll
  for (int mi = 0; mi < 2; ++mi)
#pragma unroll
    for (int t = 0; t < 4; ++t)
#pragma unroll
      for (int r = 0; r < 8; ++r)
        out[(size_t)(s0 + wr * 32 + mi * 16 + hf * 8 + r) * S_ +
            t0 + wc * 64 + t * 16 + n] = acc[mi][t][r] * scale;
}

// ---------------- Kernel 3: row softmax in place ----------------
__global__ __launch_bounds__(256) void softmax_kernel(float* __restrict__ attn) {
  __shared__ float red[256];
  float* p = attn + (size_t)blockIdx.x * S_;
  int tid = threadIdx.x;

  float v[8];
  float m = -3.402823466e+38f;
#pragma unroll
  for (int i = 0; i < 8; ++i) { v[i] = p[tid + i * 256]; m = fmaxf(m, v[i]); }
  red[tid] = m; __syncthreads();
  for (int s = 128; s > 0; s >>= 1) {
    if (tid < s) red[tid] = fmaxf(red[tid], red[tid + s]);
    __syncthreads();
  }
  m = red[0]; __syncthreads();

  float sum = 0.f;
#pragma unroll
  for (int i = 0; i < 8; ++i) { v[i] = __expf(v[i] - m); sum += v[i]; }
  red[tid] = sum; __syncthreads();
  for (int s = 128; s > 0; s >>= 1) {
    if (tid < s) red[tid] += red[tid + s];
    __syncthreads();
  }
  float inv = 1.0f / red[0];
#pragma unroll
  for (int i = 0; i < 8; ++i) p[tid + i * 256] = v[i] * inv;
}

// ---------------- Kernel 4: weighted = P @ V ----------------
// grid: b(4) x Sblk(16) x Nblk(4) = 256 blocks; K loop over S=2048
__global__ __launch_bounds__(256) void pv_kernel(
    const float* __restrict__ attn, const bf16* __restrict__ Vt,
    float* __restrict__ out) {
  __shared__ bf16 As[2][128 * LDS_STRIDE];
  __shared__ bf16 Bs[2][128 * LDS_STRIDE];

  int tid = threadIdx.x, lane = tid & 31, wv = tid >> 5;
  int wr = wv >> 1, wc = wv & 1;

  int bid = blockIdx.x;
  int b   = bid >> 6;
  int rem = bid & 63;
  int s0  = (rem >> 2) * 128;
  int n0  = (rem & 3) * 128;

  const float* P = attn + (size_t)b * S_ * S_;
  const bf16*  V = Vt   + (size_t)b * DQ_ * S_;

  v8f acc[2][4];
#pragma unroll
  for (int mi = 0; mi < 2; ++mi)
#pragma unroll
    for (int t = 0; t < 4; ++t) acc[mi][t] = v8f{};

  stage_cvt_f32(As[0], P, S_, s0, 0, tid);
  stage_async_bf16(Bs[0], V, S_, n0, 0, tid);

  int buf = 0;
  for (int k0 = 0; k0 < S_; k0 += 32) {
    wait_async();
    __syncthreads();
    int nxt = buf ^ 1;
    if (k0 + 32 < S_) {
      stage_cvt_f32(As[nxt], P, S_, s0, k0 + 32, tid);
      stage_async_bf16(Bs[nxt], V, S_, n0, k0 + 32, tid);
    }
    v16bf a0 = lds_a(As[buf], wr * 32, lane);
    v16bf a1 = lds_a(As[buf], wr * 32 + 16, lane);
#pragma unroll
    for (int t = 0; t < 4; ++t) {
      v16bf bm = lds_b(Bs[buf], wc * 64 + t * 16, lane);
      acc[0][t] = wmma_bf16(a0, bm, acc[0][t]);
      acc[1][t] = wmma_bf16(a1, bm, acc[1][t]);
    }
    __syncthreads();
    buf = nxt;
  }

  int hf = lane >> 4, n = lane & 15;
  float* o = out + (size_t)b * S_ * DQ_;
#pragma unroll
  for (int mi = 0; mi < 2; ++mi)
#pragma unroll
    for (int t = 0; t < 4; ++t)
#pragma unroll
      for (int r = 0; r < 8; ++r)
        o[(size_t)(s0 + wr * 32 + mi * 16 + hf * 8 + r) * DQ_ +
          n0 + wc * 64 + t * 16 + n] = acc[mi][t][r];
}

// ---------------- launcher ----------------
extern "C" void kernel_launch(void* const* d_in, const int* in_sizes, int n_in,
                              void* d_out, int out_size, void* d_ws, size_t ws_size,
                              hipStream_t stream) {
  const float* x  = (const float*)d_in[0];
  const float* Wq = (const float*)d_in[1];
  const float* bq = (const float*)d_in[2];
  const float* Wk = (const float*)d_in[3];
  const float* bk = (const float*)d_in[4];
  const float* Wv = (const float*)d_in[5];
  const float* bv = (const float*)d_in[6];

  float* attn     = (float*)d_out;                 // [B,S,S]
  float* weighted = attn + (size_t)B_ * S_ * S_;   // [B,S,DQ]

  bf16* Wb = (bf16*)d_ws;
  bf16* Qb = Wb + (size_t)3 * DQ_ * DI_;
  bf16* Kb = Qb + (size_t)B_ * S_ * DQ_;
  bf16* Vt = Kb + (size_t)B_ * S_ * DQ_;

  convert_w_kernel<<<dim3((DQ_ * DI_) / 256), dim3(256), 0, stream>>>(Wq, Wk, Wv, Wb);
  qkv_kernel     <<<dim3(768),  dim3(256), 0, stream>>>(x, Wb, bq, bk, bv, Qb, Kb, Vt);
  scores_kernel  <<<dim3(1024), dim3(256), 0, stream>>>(Qb, Kb, attn);
  softmax_kernel <<<dim3(B_ * S_), dim3(256), 0, stream>>>(attn);
  pv_kernel      <<<dim3(256),  dim3(256), 0, stream>>>(attn, Vt, weighted);
}
